// FastAttention_29085518529108
// MI455X (gfx1250) — matrix operand order repaired and checked by reading, not compiled, not run on hardware
//
#include <hip/hip_runtime.h>

// CDNA5 / gfx1250, wave32. All GEMMs via v_wmma_f32_16x16x32_bf16.
typedef __attribute__((ext_vector_type(16))) __bf16 v16bf;
typedef __attribute__((ext_vector_type(8)))  float  v8f;

#define BN_SCALE 0.9995003746877732f   // 1/sqrt(1+1e-3)
#define NB   8
#define NPIX 16384
#define CCH  128
#define CP   32
#define NROWS (NB * NPIX)              // 131072

__device__ __forceinline__ v8f wmma_bf16(v16bf a, v16bf b, v8f c) {
    // (neg_a, A, neg_b, B, c_mod, C, reuse_a, reuse_b)
    return __builtin_amdgcn_wmma_f32_16x16x32_bf16(false, a, false, b, (short)0, c, false, false);
}

// ---- Fragment loaders (ISA 7.12.2 layouts, wave32) ----------------------
// A 16x32 bf16: lane = m + 16h holds row m; elems 0..7 = K[8h..8h+7],
// elems 8..15 = K[16+8h .. 16+8h+7].  p points at row start + k0.
__device__ __forceinline__ v16bf load_a_bf16(const __bf16* p, int h) {
    v16bf a;
#pragma unroll
    for (int i = 0; i < 8; ++i) a[i] = p[8 * h + i];
#pragma unroll
    for (int i = 0; i < 8; ++i) a[8 + i] = p[16 + 8 * h + i];
    return a;
}
__device__ __forceinline__ v16bf load_a_f32(const float* p, int h) {
    v16bf a;
#pragma unroll
    for (int i = 0; i < 8; ++i) a[i] = (__bf16)p[8 * h + i];
#pragma unroll
    for (int i = 0; i < 8; ++i) a[8 + i] = (__bf16)p[16 + 8 * h + i];
    return a;
}
// B 32x16 bf16 from column-major ("W^T") storage wt[col][k], row stride `stride`:
// lane = c' + 16h holds column col; elems 0..7 = K[8h..8h+7], elems 8..15 = K[16+8h..16+8h+7].
__device__ __forceinline__ v16bf load_b_wt(const __bf16* wt, int stride, int col,
                                           int h, int k0) {
    const __bf16* p = wt + col * stride + k0;
    v16bf b;
#pragma unroll
    for (int i = 0; i < 8; ++i) b[i] = p[8 * h + i];
#pragma unroll
    for (int i = 0; i < 8; ++i) b[8 + i] = p[16 + 8 * h + i];
    return b;
}

// ========================================================================
// Stage 1: per-pixel projections x@[vW|qW|kW] + BN (+relu/l2norm prep).
// Writes pre_v (f32 -> d_out), q (bf16 [row][32]), kT (bf16 [b][32][n]).
// ========================================================================
__global__ __launch_bounds__(256)
void qkv_proj(const float* __restrict__ x,
              const float* __restrict__ qW, const float* __restrict__ qb,
              const float* __restrict__ qg, const float* __restrict__ qbe,
              const float* __restrict__ kW, const float* __restrict__ kb,
              const float* __restrict__ kg, const float* __restrict__ kbe,
              const float* __restrict__ vW, const float* __restrict__ vb,
              const float* __restrict__ vg, const float* __restrict__ vbe,
              float*  __restrict__ pre_v_out,   // [NROWS][128]
              __bf16* __restrict__ q_out,       // [NROWS][32]
              __bf16* __restrict__ kT_out)      // [NB][32][NPIX]
{
    __shared__ __bf16 wt[192 * 132];   // W^T: cols 0..127 vW, 128..159 qW, 160..191 kW
    const int tid = threadIdx.x;
#pragma unroll 4
    for (int i = 0; i < 96; ++i) {
        int idx = i * 256 + tid;
        int n = idx >> 7, c = idx & 127;
        float w;
        if (n < 128)      w = vW[c * 128 + n];
        else if (n < 160) w = qW[c * 32 + (n - 128)];
        else              w = kW[c * 32 + (n - 160)];
        wt[n * 132 + c] = (__bf16)w;
    }
    __syncthreads();

    const int wave = tid >> 5, lane = tid & 31;
    const int m = lane & 15, h = lane >> 4;     // m doubles as c' in D layout
    const int row0 = blockIdx.x * 128 + wave * 16;
    const int b    = row0 / NPIX;
    const int nl0  = row0 - b * NPIX;

    v16bf a[4];
#pragma unroll
    for (int kc = 0; kc < 4; ++kc)
        a[kc] = load_a_f32(x + (size_t)(row0 + m) * CCH + kc * 32, h);

    // ---- V path: 8 column tiles -> pre_v (store f32 into d_out) ----
#pragma unroll
    for (int nt = 0; nt < 8; ++nt) {
        v8f acc = {};
#pragma unroll
        for (int kc = 0; kc < 4; ++kc)
            acc = wmma_bf16(a[kc], load_b_wt(wt, 132, nt * 16 + m, h, kc * 32), acc);
        const int d = nt * 16 + m;
        const float g = vg[d] * BN_SCALE, be = vbe[d], bi = vb[d];
#pragma unroll
        for (int r = 0; r < 8; ++r) {
            const int row = row0 + r + 8 * h;
            pre_v_out[(size_t)row * CCH + d] = (acc[r] + bi) * g + be;
        }
    }

    // ---- Q path: tiles 8,9 (e = 0..31) + l2norm over the 32 channels ----
    {
        v8f a0 = {}, a1 = {};
#pragma unroll
        for (int kc = 0; kc < 4; ++kc) {
            a0 = wmma_bf16(a[kc], load_b_wt(wt, 132, 128 + m, h, kc * 32), a0);
            a1 = wmma_bf16(a[kc], load_b_wt(wt, 132, 144 + m, h, kc * 32), a1);
        }
        const int e0 = m, e1 = 16 + m;
        const float g0 = qg[e0] * BN_SCALE, b0 = qbe[e0], i0 = qb[e0];
        const float g1 = qg[e1] * BN_SCALE, b1 = qbe[e1], i1 = qb[e1];
        float q0[8], q1[8], ss[8];
#pragma unroll
        for (int r = 0; r < 8; ++r) {
            q0[r] = (a0[r] + i0) * g0 + b0;
            q1[r] = (a1[r] + i1) * g1 + b1;
            ss[r] = q0[r] * q0[r] + q1[r] * q1[r];
        }
#pragma unroll
        for (int mk = 1; mk < 16; mk <<= 1)   // reduce over the 16-lane half
#pragma unroll
            for (int r = 0; r < 8; ++r) ss[r] += __shfl_xor(ss[r], mk, 32);
#pragma unroll
        for (int r = 0; r < 8; ++r) {
            const float rn = rsqrtf(fmaxf(ss[r], 1e-12f));
            const int row = row0 + r + 8 * h;
            q_out[(size_t)row * CP + e0] = (__bf16)(q0[r] * rn);
            q_out[(size_t)row * CP + e1] = (__bf16)(q1[r] * rn);
        }
    }

    // ---- K path: tiles 10,11 + l2norm, stored transposed [b][e][n] ----
    {
        v8f a0 = {}, a1 = {};
#pragma unroll
        for (int kc = 0; kc < 4; ++kc) {
            a0 = wmma_bf16(a[kc], load_b_wt(wt, 132, 160 + m, h, kc * 32), a0);
            a1 = wmma_bf16(a[kc], load_b_wt(wt, 132, 176 + m, h, kc * 32), a1);
        }
        const int e0 = m, e1 = 16 + m;
        const float g0 = kg[e0] * BN_SCALE, b0 = kbe[e0], i0 = kb[e0];
        const float g1 = kg[e1] * BN_SCALE, b1 = kbe[e1], i1 = kb[e1];
        float k0[8], k1[8], ss[8];
#pragma unroll
        for (int r = 0; r < 8; ++r) {
            k0[r] = (a0[r] + i0) * g0 + b0;
            k1[r] = (a1[r] + i1) * g1 + b1;
            ss[r] = k0[r] * k0[r] + k1[r] * k1[r];
        }
#pragma unroll
        for (int mk = 1; mk < 16; mk <<= 1)
#pragma unroll
            for (int r = 0; r < 8; ++r) ss[r] += __shfl_xor(ss[r], mk, 32);
#pragma unroll
        for (int r = 0; r < 8; ++r) {
            const float rn = rsqrtf(fmaxf(ss[r], 1e-12f));
            const int nl = nl0 + r + 8 * h;
            kT_out[((size_t)b * CP + e0) * NPIX + nl] = (__bf16)(k0[r] * rn);
            kT_out[((size_t)b * CP + e1) * NPIX + nl] = (__bf16)(k1[r] * rn);
        }
    }
}

// ========================================================================
// Stage 2: kTv split-K partials. Grid = NB*64 blocks; each covers 256 n.
// v = relu(pre_v) staged TRANSPOSED in LDS (vT[c][nl]) so B fragments are
// two runs of 8 contiguous bf16 per lane (ds_load_2addr_b64 pairs, no u16 gathers).
// Double-buffered -> one barrier per K-chunk; next chunk prefetched.
// Deterministic (split-K partials, no atomics).
// ========================================================================
__global__ __launch_bounds__(256)
void kv_outer(const float* __restrict__ pre_v, const __bf16* __restrict__ kT,
              float* __restrict__ part)   // [NB][64][32][128]
{
    __shared__ __bf16 vT[2][128 * 36];   // [buf][c][nl], nl-chunk of 32, padded
    const int blk = blockIdx.x;
    const int b = blk >> 6, s = blk & 63;
    const int nbase = s * 256;
    const int tid = threadIdx.x, wave = tid >> 5, lane = tid & 31;
    const int m = lane & 15, h = lane >> 4;
    const int etile = wave >> 2, cg = wave & 3;
    const int c0 = (cg * 2 + 0) * 16 + m, c1 = c0 + 16;
    const int e  = etile * 16 + m;

    const float*  pv  = pre_v + ((size_t)b * NPIX + nbase) * CCH;
    const __bf16* kTe = kT + ((size_t)b * CP + e) * NPIX + nbase;
    v8f acc0 = {}, acc1 = {};

    for (int it = 0; it < 8; ++it) {
        __bf16* buf = vT[it & 1];
#pragma unroll 4
        for (int i = 0; i < 16; ++i) {
            int idx = i * 256 + tid;
            int nl = idx >> 7, c = idx & 127;
            float pz = pv[(size_t)(it * 32 + nl) * CCH + c];
            buf[c * 36 + nl] = (__bf16)fmaxf(pz, 0.f);   // transposed store
        }
        if (it + 1 < 8)   // pull next 16KB chunk toward the caches
            __builtin_prefetch(pv + (size_t)(it + 1) * 32 * CCH + tid * 16, 0, 0);
        __syncthreads();   // fill(buf) -> compute(buf); alternate buffer
                           // makes the next fill race-free without a 2nd sync
        v16bf af = load_a_bf16(kTe + it * 32, h);
        acc0 = wmma_bf16(af, load_b_wt(buf, 36, c0, h, 0), acc0);
        acc1 = wmma_bf16(af, load_b_wt(buf, 36, c1, h, 0), acc1);
    }
    float* dst = part + (((size_t)b * 64 + s) * CP) * CCH;
#pragma unroll
    for (int r = 0; r < 8; ++r) {
        const int er = etile * 16 + r + 8 * h;
        dst[(size_t)er * CCH + c0] = acc0[r];
        dst[(size_t)er * CCH + c1] = acc1[r];
    }
}

// ========================================================================
// Stage 2.5: fixed-order split-K reduction + tiny M = kTv @ oW per batch.
// Stored d-major (M^T [d][e]) so stage-3 B fragments are contiguous.
// ========================================================================
__global__ __launch_bounds__(256)
void reduce_project(const float* __restrict__ part, const float* __restrict__ oW,
                    float* __restrict__ MT)   // [NB][128][32]
{
    __shared__ float ktv[32 * 132];
    const int b = blockIdx.x, tid = threadIdx.x;
#pragma unroll 4
    for (int i = 0; i < 16; ++i) {
        int idx = i * 256 + tid;
        int e = idx >> 7, c = idx & 127;
        const float* p = part + (((size_t)b * 64) * CP + e) * CCH + c;
        float sum = 0.f;
        for (int s = 0; s < 64; ++s) sum += p[(size_t)s * CP * CCH];
        ktv[e * 132 + c] = sum;
    }
    __syncthreads();
#pragma unroll 4
    for (int i = 0; i < 16; ++i) {
        int idx = i * 256 + tid;
        int d = idx >> 5, e = idx & 31;
        float sum = 0.f;
        for (int c = 0; c < CCH; ++c) sum += ktv[e * 132 + c] * oW[c * CCH + d];
        MT[((size_t)b * CCH + d) * CP + e] = sum;
    }
}

// ========================================================================
// Stage 3: out = pre_v + BN(q @ M + ob). One K=32 WMMA per 16x16 tile.
// ========================================================================
__global__ __launch_bounds__(256)
void attn_out(const __bf16* __restrict__ q, const float* __restrict__ MT,
              const float* __restrict__ ob, const float* __restrict__ og,
              const float* __restrict__ obe,
              float* __restrict__ out)   // d_out: already holds pre_v
{
    __shared__ __bf16 mt[128 * 36];    // M^T bf16, padded rows
    const int tid = threadIdx.x;
    const int row_blk = blockIdx.x * 128;
    const int b = row_blk / NPIX;
#pragma unroll 4
    for (int i = 0; i < 16; ++i) {
        int idx = i * 256 + tid;
        int d = idx >> 5, e = idx & 31;
        mt[d * 36 + e] = (__bf16)MT[((size_t)b * CCH + d) * CP + e];
    }
    __syncthreads();

    const int wave = tid >> 5, lane = tid & 31;
    const int m = lane & 15, h = lane >> 4;
    const int row0 = row_blk + wave * 16;

    v16bf aq = load_a_bf16(q + (size_t)(row0 + m) * CP, h);
#pragma unroll
    for (int dt = 0; dt < 8; ++dt) {
        v8f acc = {};
        acc = wmma_bf16(aq, load_b_wt(mt, 36, dt * 16 + m, h, 0), acc);
        const int d = dt * 16 + m;
        const float g = og[d] * BN_SCALE, be = obe[d], bi = ob[d];
#pragma unroll
        for (int r = 0; r < 8; ++r) {
            const size_t o = (size_t)(row0 + r + 8 * h) * CCH + d;
            out[o] = out[o] + (acc[r] + bi) * g + be;
        }
    }
}

// ========================================================================
extern "C" void kernel_launch(void* const* d_in, const int* in_sizes, int n_in,
                              void* d_out, int out_size, void* d_ws, size_t ws_size,
                              hipStream_t stream) {
    const float* x   = (const float*)d_in[0];
    const float* qW  = (const float*)d_in[1];
    const float* qb  = (const float*)d_in[2];
    const float* qg  = (const float*)d_in[3];
    const float* qbe = (const float*)d_in[4];
    const float* kW  = (const float*)d_in[5];
    const float* kb  = (const float*)d_in[6];
    const float* kg  = (const float*)d_in[7];
    const float* kbe = (const float*)d_in[8];
    const float* vW  = (const float*)d_in[9];
    const float* vb  = (const float*)d_in[10];
    const float* vg  = (const float*)d_in[11];
    const float* vbe = (const float*)d_in[12];
    const float* oW  = (const float*)d_in[13];
    const float* ob  = (const float*)d_in[14];
    const float* og  = (const float*)d_in[15];
    const float* obe = (const float*)d_in[16];
    float* out = (float*)d_out;

    char* ws = (char*)d_ws;
    __bf16* wq   = (__bf16*)(ws);                        // 8 MiB  q  [NROWS][32]
    __bf16* wkT  = (__bf16*)(ws + ((size_t)8 << 20));    // 8 MiB  kT [NB][32][NPIX]
    float*  wprt = (float*) (ws + ((size_t)16 << 20));   // 8 MiB  partials [NB][64][32][128]
    float*  wMT  = (float*) (ws + ((size_t)24 << 20));   // 128 KiB M^T [NB][128][32]

    qkv_proj<<<NROWS / 128, 256, 0, stream>>>(x, qW, qb, qg, qbe, kW, kb, kg, kbe,
                                              vW, vb, vg, vbe, out, wq, wkT);
    kv_outer<<<NB * 64, 256, 0, stream>>>(out, wkT, wprt);
    reduce_project<<<NB, 256, 0, stream>>>(wprt, oW, wMT);
    attn_out<<<NROWS / 128, 256, 0, stream>>>(wq, wMT, ob, og, obe, out);
}
